// TinyLSTM_26276609917046
// MI455X (gfx1250) — matrix operand order repaired and checked by reading, not compiled, not run on hardware
//
#include <hip/hip_runtime.h>
#include <stdint.h>

// TinyLSTM on MI455X (gfx1250, wave32, WMMA).
//
// Key transforms vs reference:
//  1. xg = emb[x] @ W_ih.T + b  ==> 100x128 token->gate-bias table in LDS (51KB).
//  2. Recurrence per step computed as W_hh[128x32] x h^T[32x16] with
//     v_wmma_f32_16x16x32_f16 (K=32 == HIDDEN, one WMMA per 16x16 gate tile,
//     8 WMMAs/step). W_hh stays resident in registers as A-fragments.
//  3. h D-layout -> next-step B-layout transpose done with packed-f16 +
//     8 lane-half shuffles (no LDS traffic in the hot loop except the
//     contiguous ds_load_b128 gathers of the token gate-bias row).

#define VOCABSZ 100
#define EMBEDSZ 16
#define HIDDENSZ 32
#define NGATE   128   // 4*HIDDEN
#define TSTEPS  256

typedef _Float16 v16h __attribute__((ext_vector_type(16)));
typedef float    v8f  __attribute__((ext_vector_type(8)));
typedef float    v4f  __attribute__((ext_vector_type(4)));
typedef uint32_t v8u  __attribute__((ext_vector_type(8)));

static __device__ __forceinline__ uint32_t pkh2(float a, float b) {
  union { _Float16 h[2]; uint32_t u; } r;
  r.h[0] = (_Float16)a;   // bits [15:0]
  r.h[1] = (_Float16)b;   // bits [31:16]
  return r.u;
}

static __device__ __forceinline__ float fast_tanh(float x) {
#if __has_builtin(__builtin_amdgcn_tanhf)
  return __builtin_amdgcn_tanhf(x);            // native v_tanh_f32 on gfx1250
#elif __has_builtin(__builtin_amdgcn_tanh_f32)
  return __builtin_amdgcn_tanh_f32(x);
#else
  // tanh(x) = 2/(1+2^(-2x*log2e)) - 1  -> v_mul, v_exp, v_add, v_rcp, v_fma
  float e = __builtin_amdgcn_exp2f(-2.8853900817779268f * x);
  return 2.0f * __builtin_amdgcn_rcpf(1.0f + e) - 1.0f;
#endif
}

static __device__ __forceinline__ float fast_sigmoid(float x) {
  return 0.5f + 0.5f * fast_tanh(0.5f * x);
}

__global__ __launch_bounds__(64) void TinyLSTM_gfx1250_kernel(
    const int*   __restrict__ x,      // [B, T]
    const float* __restrict__ emb,    // [100, 16]
    const float* __restrict__ W_ih,   // [128, 16]
    const float* __restrict__ W_hh,   // [128, 32]
    const float* __restrict__ b_ih,   // [128]
    const float* __restrict__ b_hh,   // [128]
    const float* __restrict__ fc_w,   // [2, 32]
    const float* __restrict__ fc_b,   // [2]
    float*       __restrict__ out)    // [B, 2]
{
  // Token -> (xg row) table: G[v][g] = dot(emb[v], W_ih[g]) + b_ih[g] + b_hh[g]
  __shared__ float Gt[VOCABSZ * NGATE];   // 51,200 bytes

  const int tid = threadIdx.x;
  for (int idx = tid; idx < VOCABSZ * NGATE; idx += 64) {
    const int v = idx >> 7;
    const int g = idx & (NGATE - 1);
    const float* er = emb  + v * EMBEDSZ;
    const float* wr = W_ih + g * EMBEDSZ;
    float acc = b_ih[g] + b_hh[g];
#pragma unroll
    for (int e = 0; e < EMBEDSZ; ++e) acc += er[e] * wr[e];
    Gt[idx] = acc;
  }
  __syncthreads();

  const int lane = tid & 31;
  const int wave = tid >> 5;
  const int hi   = lane >> 4;        // 0 = lanes 0-15, 1 = lanes 16-31
  const int col  = lane & 15;        // batch column within tile; also A-row M
  const int k0   = hi * 8;           // C/D gate-row half offset; A reads K from 2*k0 (contiguous)
  const int b0   = (blockIdx.x * 2 + wave) * 16;   // first batch row of tile

  // ---- W_hh as 8 resident A-fragments (16x32 f16 each) --------------------
  // A layout (16-bit, 16x32): lane m%16 holds row M; halves 0..15 = K
  // 16*hi+0..15 (contiguous, matching the B fragment built below).
  v16h a[8];
#pragma unroll
  for (int t8 = 0; t8 < 8; ++t8) {
    const float* wr = W_hh + (16 * t8 + col) * HIDDENSZ + 2 * k0;
#pragma unroll
    for (int j = 0; j < 8; ++j) {
      a[t8][j]     = (_Float16)wr[j];
      a[t8][j + 8] = (_Float16)wr[8 + j];
    }
  }

  // ---- state --------------------------------------------------------------
  v8u bu  = {0, 0, 0, 0, 0, 0, 0, 0};              // B fragment (h^T), h0 = 0
  v8f cs0 = {0, 0, 0, 0, 0, 0, 0, 0};              // c, hidden 8*hi+v
  v8f cs1 = {0, 0, 0, 0, 0, 0, 0, 0};              // c, hidden 16+8*hi+v
  v8f hs0 = {0, 0, 0, 0, 0, 0, 0, 0};
  v8f hs1 = {0, 0, 0, 0, 0, 0, 0, 0};

  const int* xrow = x + (b0 + col) * TSTEPS;
  int tok = xrow[0];

  for (int t = 0; t < TSTEPS; ++t) {
    const int tok_nxt = (t + 1 < TSTEPS) ? xrow[t + 1] : 0;  // prefetch
    const float* grow = &Gt[tok * NGATE + k0];   // per-lane gate-bias row

    const v16h b = __builtin_bit_cast(v16h, bu);
    v8f d[8];
#pragma unroll
    for (int tt = 0; tt < 8; ++tt) {
      // C/D layout: VGPR v -> gate row 16*tt + 8*hi + v; so the 8 values a
      // lane needs are 8 consecutive floats of G[tok] -> 2x ds_load_b128.
      const v4f* gp = (const v4f*)(grow + 16 * tt);
      const v4f glo = gp[0];
      const v4f ghi = gp[1];
      v8f c;
      c[0] = glo[0]; c[1] = glo[1]; c[2] = glo[2]; c[3] = glo[3];
      c[4] = ghi[0]; c[5] = ghi[1]; c[6] = ghi[2]; c[7] = ghi[3];
      d[tt] = __builtin_amdgcn_wmma_f32_16x16x32_f16(
          false, a[tt], false, b, (short)0, c, false, false);
    }

    // Gate tiles: i = d[0..1], f = d[2..3], g = d[4..5], o = d[6..7]
#pragma unroll
    for (int v = 0; v < 8; ++v) {
      float I0 = fast_sigmoid(d[0][v]);
      float F0 = fast_sigmoid(d[2][v]);
      float G0 = fast_tanh  (d[4][v]);
      float O0 = fast_sigmoid(d[6][v]);
      float c0 = F0 * cs0[v] + I0 * G0;
      cs0[v] = c0;
      hs0[v] = O0 * fast_tanh(c0);

      float I1 = fast_sigmoid(d[1][v]);
      float F1 = fast_sigmoid(d[3][v]);
      float G1 = fast_tanh  (d[5][v]);
      float O1 = fast_sigmoid(d[7][v]);
      float c1 = F1 * cs1[v] + I1 * G1;
      cs1[v] = c1;
      hs1[v] = O1 * fast_tanh(c1);
    }

    // ---- h (D layout) -> B fragment (K=hidden, N=batch) in registers ------
    // B layout (16-bit, 32x16): dword j = K {2j,2j+1} for lanes 0-15,
    // K {16+2j,17+2j} for lanes 16-31, N = lane%16. Lane l and l^16 share
    // the same batch column, so one half-swap shuffle completes the K range.
    uint32_t pk[8], sh[8];
#pragma unroll
    for (int j = 0; j < 4; ++j) {
      pk[j]     = pkh2(hs0[2 * j], hs0[2 * j + 1]);   // hidden 8*hi + {2j,2j+1}
      pk[j + 4] = pkh2(hs1[2 * j], hs1[2 * j + 1]);   // hidden 16+8*hi + ...
    }
#pragma unroll
    for (int j = 0; j < 8; ++j)
      sh[j] = (uint32_t)__shfl_xor((int)pk[j], 16, 32);
#pragma unroll
    for (int j = 0; j < 4; ++j) {
      bu[j]     = hi ? sh[j + 4] : pk[j];
      bu[j + 4] = hi ? pk[j + 4] : sh[j];
    }

    tok = tok_nxt;
  }

  // ---- FC head: out[b] = h_final @ fc_w.T + fc_b (OUT=2) ------------------
  float p0 = 0.0f, p1 = 0.0f;
#pragma unroll
  for (int v = 0; v < 8; ++v) {
    const int h0i = 8 * hi + v;
    const int h1i = 16 + 8 * hi + v;
    p0 += hs0[v] * fc_w[h0i]      + hs1[v] * fc_w[h1i];
    p1 += hs0[v] * fc_w[32 + h0i] + hs1[v] * fc_w[32 + h1i];
  }
  p0 += __shfl_xor(p0, 16, 32);   // combine complementary hidden halves
  p1 += __shfl_xor(p1, 16, 32);
  if (hi == 0) {
    out[(b0 + col) * 2 + 0] = p0 + fc_b[0];
    out[(b0 + col) * 2 + 1] = p1 + fc_b[1];
  }
}

extern "C" void kernel_launch(void* const* d_in, const int* in_sizes, int n_in,
                              void* d_out, int out_size, void* d_ws, size_t ws_size,
                              hipStream_t stream) {
  const int*   x    = (const int*)  d_in[0];
  const float* emb  = (const float*)d_in[1];
  const float* W_ih = (const float*)d_in[2];
  const float* W_hh = (const float*)d_in[3];
  const float* b_ih = (const float*)d_in[4];
  const float* b_hh = (const float*)d_in[5];
  const float* fc_w = (const float*)d_in[6];
  const float* fc_b = (const float*)d_in[7];
  float* out = (float*)d_out;

  const int B = in_sizes[0] / TSTEPS;   // 4096
  const int blocks = B / 32;            // 2 waves/block, 16 batch rows/wave
  TinyLSTM_gfx1250_kernel<<<dim3(blocks), dim3(64), 0, stream>>>(
      x, emb, W_ih, W_hh, b_ih, b_hh, fc_w, fc_b, out);
}
